// DecoderLayer_13683765805857
// MI455X (gfx1250) — matrix operand order repaired and checked by reading, not compiled, not run on hardware
//
#include <hip/hip_runtime.h>
#include <hip/hip_bf16.h>

// ---------------- problem constants ----------------
#define B_     2
#define S_     2048
#define D_     1024
#define H_     16
#define HD_    64
#define FF_    4096
#define BS_    (B_ * S_)

typedef __attribute__((ext_vector_type(16))) __bf16 bf16x16;
typedef __attribute__((ext_vector_type(8)))  float  f32x8;
typedef unsigned int u32;
typedef __attribute__((ext_vector_type(4))) u32 u32x4;
typedef __attribute__((ext_vector_type(8))) u32 u32x8;

union Frag {
    bf16x16 v;
    uint4   u[2];
};

// =============================================================
// Tensor Data Mover: async 2D tile load Global -> LDS.
// D# layout per CDNA5 ISA.
// Tile: tile_d1 rows x tile_d0 elements (2-byte), row stride = stride0.
// LDS destination rows are padded +16B per 64B (pad_interval=16 DWORDs,
// pad_amount=4 DWORDs) to match the [128][40] bf16 LDS tiles.
// =============================================================
__device__ __forceinline__ void tdm_load_2d(
    u32 lds_addr, const void* gaddr,
    u32 tensor_d0, u32 tensor_d1,
    u32 tile_d0, u32 tile_d1, u32 stride0)
{
    unsigned long long ga = (unsigned long long)(size_t)gaddr;
    u32x4 g0;
    g0[0] = 1u;                                     // count=1, user descriptor
    g0[1] = lds_addr;                               // lds_addr[31:0]
    g0[2] = (u32)ga;                                // global_addr[31:0]
    g0[3] = (u32)((ga >> 32) & 0x1FFFFFFull)        // global_addr[56:32]
          | (2u << 30);                             // type = 2 ("image")
    u32x8 g1;
    g1[0] = (1u << 16)                              // data_size = 2 bytes
          | (1u << 20)                              // pad_enable
          | (3u << 22)                              // pad_interval: 16 DWORDs
          | (3u << 25);                             // pad_amount: 4 DWORDs
    g1[1] = (tensor_d0 & 0xFFFFu) << 16;            // tensor_dim0[15:0]
    g1[2] = (tensor_d0 >> 16) | ((tensor_d1 & 0xFFFFu) << 16);
    g1[3] = (tensor_d1 >> 16) | (tile_d0 << 16);    // tile_dim0
    g1[4] = tile_d1;                                // tile_dim1 (tile_dim2 = 0)
    g1[5] = stride0;                                // tensor_dim0_stride[31:0]
    g1[6] = 0;                                      // stride0[47:32], stride1[15:0]
    g1[7] = 0;
    asm volatile("tensor_load_to_lds %0, %1" :: "s"(g0), "s"(g1) : "memory");
}

// =============================================================
// Weight conversion kernels (fp32 -> bf16, into WMMA Bt layout)
// =============================================================

// Per-head weight [H, D, HD] -> Bt layout [N=H*HD][K=D]
__global__ __launch_bounds__(256) void cvt_perhead_kernel(
    const float* __restrict__ in, __bf16* __restrict__ out)
{
    size_t idx = (size_t)blockIdx.x * 256 + threadIdx.x;  // n*D + d
    if (idx >= (size_t)D_ * D_) return;
    int n = (int)(idx / D_), d = (int)(idx % D_);
    int h = n / HD_, j = n % HD_;
    out[idx] = (__bf16)in[((size_t)h * D_ + d) * HD_ + j];
}

// Dense weight [K, N] -> Bt layout [N][K]
__global__ __launch_bounds__(256) void cvt_T_kernel(
    const float* __restrict__ in, __bf16* __restrict__ out, int K, int N)
{
    size_t idx = (size_t)blockIdx.x * 256 + threadIdx.x;  // n*K + k
    if (idx >= (size_t)K * N) return;
    size_t n = idx / K, k = idx % K;
    out[idx] = (__bf16)in[k * (size_t)N + n];
}

// v [BS, D] (bf16, per-head [b,s,h,j]) -> vT [B,H,HD,S]
__global__ __launch_bounds__(256) void transpose_v_kernel(
    const __bf16* __restrict__ v, __bf16* __restrict__ vT)
{
    size_t idx = (size_t)blockIdx.x * 256 + threadIdx.x;
    if (idx >= (size_t)BS_ * D_) return;
    int s = (int)(idx % S_); size_t t = idx / S_;
    int j = (int)(t % HD_);  t /= HD_;
    int h = (int)(t % H_);   int b = (int)(t / H_);
    vT[idx] = v[((size_t)b * S_ + s) * D_ + h * HD_ + j];
}

// =============================================================
// LayerNorm: fp32 rows -> bf16 rows
// =============================================================
__global__ __launch_bounds__(256) void ln_kernel(
    const float* __restrict__ x, const float* __restrict__ g,
    const float* __restrict__ beta, __bf16* __restrict__ out)
{
    const int row = blockIdx.x;
    const int tid = threadIdx.x;
    const float* xr = x + (size_t)row * D_;

    float v[4], s = 0.f, s2 = 0.f;
    for (int i = 0; i < 4; ++i) {
        v[i] = xr[tid + i * 256];
        s += v[i];
        s2 += v[i] * v[i];
    }
    for (int off = 16; off > 0; off >>= 1) {   // wave32 reduce
        s  += __shfl_xor(s,  off, 32);
        s2 += __shfl_xor(s2, off, 32);
    }
    __shared__ float rs[8], rs2[8];
    __shared__ float smu, srstd;
    int wave = tid >> 5, lane = tid & 31;
    if (lane == 0) { rs[wave] = s; rs2[wave] = s2; }
    __syncthreads();
    if (tid == 0) {
        float S = 0.f, S2 = 0.f;
        for (int i = 0; i < 8; ++i) { S += rs[i]; S2 += rs2[i]; }
        float mu = S * (1.0f / D_);
        float var = S2 * (1.0f / D_) - mu * mu;
        smu = mu;
        srstd = rsqrtf(var + 1e-5f);
    }
    __syncthreads();
    float mu = smu, rstd = srstd;
    for (int i = 0; i < 4; ++i) {
        int c = tid + i * 256;
        float nv = (v[i] - mu) * rstd * g[c] + beta[c];
        out[(size_t)row * D_ + c] = (__bf16)nv;
    }
}

// =============================================================
// WMMA GEMM: C[M,N] = A[M,K](bf16,row-major) * Bt[N,K](bf16) + bias
// Block tile 128x128, K-step 32, 8 waves arranged 4(M) x 2(N).
// Tiles staged by the Tensor Data Mover (double-buffered): wave 0
// issues 2 TDM descriptors per K-step for the *next* tile while all
// waves run WMMAs on the current one; s_wait_tensorcnt(2) ensures the
// current tile has landed (TDM completes in-order per wave).
// =============================================================
template <bool RELU, bool ADD_RESID, bool OUT_BF16>
__global__ __launch_bounds__(256) void gemm_kernel(
    const __bf16* __restrict__ A, const __bf16* __restrict__ Bt,
    const float* __restrict__ bias, const float* __restrict__ resid,
    void* __restrict__ outv, int M, int N, int K)
{
    __shared__ __align__(16) __bf16 Al[2][128][40];  // 64B data + 16B TDM pad
    __shared__ __align__(16) __bf16 Bl[2][128][40];

    const int tid = threadIdx.x;
    const int wave = tid >> 5, lane = tid & 31;
    const int l16 = lane & 15, hi = lane >> 4;
    const int wm = wave >> 1, wn = wave & 1;  // 4 x 2 wave grid
    const int n0 = blockIdx.x * 128, m0 = blockIdx.y * 128;

    // scalar-uniform leader test -> s_cbranch (TDM ignores EXEC, so the
    // issue must be skipped by a *scalar* branch in waves 1..7)
    const bool leader = (__builtin_amdgcn_readfirstlane(wave) == 0);

    f32x8 acc[2][4] = {};
    const int ksteps = K >> 5;

    if (leader) {  // prologue: tile 0 into buffer 0
        tdm_load_2d((u32)(size_t)&Al[0][0][0], A + (size_t)m0 * K,
                    (u32)K, (u32)M, 32u, 128u, (u32)K);
        tdm_load_2d((u32)(size_t)&Bl[0][0][0], Bt + (size_t)n0 * K,
                    (u32)K, (u32)N, 32u, 128u, (u32)K);
    }

    for (int kt = 0; kt < ksteps; ++kt) {
        const int cur = kt & 1;
        if (leader) {
            if (kt + 1 < ksteps) {
                const int kb = (kt + 1) << 5;
                tdm_load_2d((u32)(size_t)&Al[cur ^ 1][0][0],
                            A + (size_t)m0 * K + kb, (u32)K, (u32)M,
                            32u, 128u, (u32)K);
                tdm_load_2d((u32)(size_t)&Bl[cur ^ 1][0][0],
                            Bt + (size_t)n0 * K + kb, (u32)K, (u32)N,
                            32u, 128u, (u32)K);
                __builtin_amdgcn_s_wait_tensorcnt(2);  // tile kt landed
            } else {
                __builtin_amdgcn_s_wait_tensorcnt(0);
            }
        }
        __syncthreads();  // tile kt visible to all waves

        Frag af[2], bfr[4];
        for (int mf = 0; mf < 2; ++mf) {
            // A frag: lanes0-15 K={0..7,16..23}; lanes16-31 K={8..15,24..31}
            const __bf16* p = &Al[cur][wm * 32 + mf * 16 + l16][hi * 8];
            af[mf].u[0] = *(const uint4*)p;
            af[mf].u[1] = *(const uint4*)(p + 16);
        }
        for (int nf = 0; nf < 4; ++nf) {
            // B frag: lanes0-15 K={0..7,16..23}; lanes16-31 K={8..15,24..31}
            const __bf16* p = &Bl[cur][wn * 64 + nf * 16 + l16][hi * 8];
            bfr[nf].u[0] = *(const uint4*)p;
            bfr[nf].u[1] = *(const uint4*)(p + 16);
        }
        for (int mf = 0; mf < 2; ++mf)
            for (int nf = 0; nf < 4; ++nf)
                acc[mf][nf] = __builtin_amdgcn_wmma_f32_16x16x32_bf16(
                    false, af[mf].v, false, bfr[nf].v, (short)0, acc[mf][nf],
                    false, false);
        __syncthreads();  // compute done; next iter may overwrite buf cur
    }

    // Epilogue. C layout: lane=(N within 16-tile), VGPR r = row r (+8 for hi)
    for (int mf = 0; mf < 2; ++mf)
        for (int nf = 0; nf < 4; ++nf) {
            int col = n0 + wn * 64 + nf * 16 + l16;
            float bc = bias ? bias[col] : 0.f;
            for (int r = 0; r < 8; ++r) {
                int row = m0 + wm * 32 + mf * 16 + hi * 8 + r;
                float vl = acc[mf][nf][r] + bc;
                if (RELU) vl = fmaxf(vl, 0.f);
                if (ADD_RESID) vl += resid[(size_t)row * N + col];
                if (OUT_BF16)
                    ((__bf16*)outv)[(size_t)row * N + col] = (__bf16)vl;
                else
                    ((float*)outv)[(size_t)row * N + col] = vl;
            }
        }
}

// =============================================================
// Flash attention. One block = one (b, h, 128-query tile).
// 8 waves x 16 query rows. Key/value tiles of 64.
// scores = Q Kt (K already in Bt layout), P V uses pre-transposed V.
// =============================================================
__global__ __launch_bounds__(256) void attn_kernel(
    const __bf16* __restrict__ q, const __bf16* __restrict__ k,
    const __bf16* __restrict__ vT, const int* __restrict__ mask,
    __bf16* __restrict__ ctx)
{
    const int qt = blockIdx.x, h = blockIdx.y, b = blockIdx.z;
    const int tid = threadIdx.x;
    const int wave = tid >> 5, lane = tid & 31;
    const int l16 = lane & 15, hi = lane >> 4;

    __shared__ __align__(16) __bf16 Klds[64][72];
    __shared__ __align__(16) __bf16 Vlds[64][72];
    __shared__ __align__(16) __bf16 Plds[8][16][72];
    __shared__ float maskAdd[64];

    const int qrow0 = qt * 128 + wave * 16;
    const __bf16* qbase = q + ((size_t)b * S_) * D_ + h * HD_;
    const __bf16* kbase = k + ((size_t)b * S_) * D_ + h * HD_;
    const __bf16* vbase = vT + (((size_t)b * H_ + h) * HD_) * S_;
    const int* mbase = mask + (size_t)b * S_;

    // Preload Q fragments (2 frags cover HD=64), kept in registers.
    Frag aq[2];
    for (int f = 0; f < 2; ++f) {
        const __bf16* p = qbase + (size_t)(qrow0 + l16) * D_ + f * 32 + hi * 8;
        aq[f].u[0] = *(const uint4*)p;
        aq[f].u[1] = *(const uint4*)(p + 16);
    }

    f32x8 acc[4] = {};
    float mrun[8], lrun[8];
    for (int r = 0; r < 8; ++r) { mrun[r] = -3.0e38f; lrun[r] = 0.f; }

    for (int kt = 0; kt < S_ / 64; ++kt) {
        // load K tile [64 t][64 hd] and Vt tile [64 hd][64 t]
        for (int u = tid; u < 512; u += 256) {
            int r = u >> 3, c = u & 7;
            *(uint4*)&Klds[r][c * 8] =
                *(const uint4*)(kbase + (size_t)(kt * 64 + r) * D_ + c * 8);
            *(uint4*)&Vlds[r][c * 8] =
                *(const uint4*)(vbase + (size_t)r * S_ + kt * 64 + c * 8);
        }
        if (kt + 1 < S_ / 64) {  // gfx1250 global_prefetch_b8 of next tiles
            __builtin_prefetch(kbase + (size_t)((kt + 1) * 64 + (tid >> 2)) * D_, 0, 1);
            __builtin_prefetch(vbase + (size_t)(tid >> 2) * S_ + (kt + 1) * 64, 0, 1);
        }
        if (tid < 64)
            maskAdd[tid] = (mbase[kt * 64 + tid] == 0) ? -1e9f : 0.0f;
        __syncthreads();

        // scores = Q @ K^T   (K tile is Bt layout: rows=t, cols=hd)
        f32x8 sacc[4] = {};
        for (int kk = 0; kk < 2; ++kk)
            for (int nf = 0; nf < 4; ++nf) {
                Frag bk;
                const __bf16* p = &Klds[nf * 16 + l16][kk * 32 + hi * 8];
                bk.u[0] = *(const uint4*)p;
                bk.u[1] = *(const uint4*)(p + 16);
                sacc[nf] = __builtin_amdgcn_wmma_f32_16x16x32_bf16(
                    false, aq[kk].v, false, bk.v, (short)0, sacc[nf],
                    false, false);
            }

        // scale + key mask + online softmax
        float tmax[8];
        for (int r = 0; r < 8; ++r) tmax[r] = -3.0e38f;
        for (int nf = 0; nf < 4; ++nf) {
            float ma = maskAdd[nf * 16 + l16];
            for (int r = 0; r < 8; ++r) {
                float vl = sacc[nf][r] * 0.125f + ma;  // 1/sqrt(64)
                sacc[nf][r] = vl;
                tmax[r] = fmaxf(tmax[r], vl);
            }
        }
        for (int off = 1; off < 16; off <<= 1)
            for (int r = 0; r < 8; ++r)
                tmax[r] = fmaxf(tmax[r], __shfl_xor(tmax[r], off, 32));

        float tsum[8];
        for (int r = 0; r < 8; ++r) {
            float mnew = fmaxf(mrun[r], tmax[r]);
            float corr = __expf(mrun[r] - mnew);
            mrun[r] = mnew;
            lrun[r] *= corr;
            for (int nf = 0; nf < 4; ++nf) acc[nf][r] *= corr;
            tsum[r] = 0.f;
        }
        for (int nf = 0; nf < 4; ++nf)
            for (int r = 0; r < 8; ++r) {
                float e = __expf(sacc[nf][r] - mrun[r]);
                sacc[nf][r] = e;
                tsum[r] += e;
            }
        for (int off = 1; off < 16; off <<= 1)
            for (int r = 0; r < 8; ++r)
                tsum[r] += __shfl_xor(tsum[r], off, 32);
        for (int r = 0; r < 8; ++r) lrun[r] += tsum[r];

        // P (C layout) -> bf16 A layout via per-wave LDS region
        for (int nf = 0; nf < 4; ++nf)
            for (int r = 0; r < 8; ++r)
                Plds[wave][hi * 8 + r][nf * 16 + l16] = (__bf16)sacc[nf][r];
        // same-wave LDS: in-order, no barrier needed

        // acc += P @ V   (Vt tile is Bt layout: rows=hd, cols=t)
        for (int kk = 0; kk < 2; ++kk) {
            Frag pa;
            const __bf16* pp = &Plds[wave][l16][kk * 32 + hi * 8];
            pa.u[0] = *(const uint4*)pp;
            pa.u[1] = *(const uint4*)(pp + 16);
            for (int nf = 0; nf < 4; ++nf) {
                Frag bv;
                const __bf16* vp = &Vlds[nf * 16 + l16][kk * 32 + hi * 8];
                bv.u[0] = *(const uint4*)vp;
                bv.u[1] = *(const uint4*)(vp + 16);
                acc[nf] = __builtin_amdgcn_wmma_f32_16x16x32_bf16(
                    false, pa.v, false, bv.v, (short)0, acc[nf], false, false);
            }
        }
        __syncthreads();
    }

    // write ctx (bf16) back as [b, s, h*HD + j]
    __bf16* cbase = ctx + ((size_t)b * S_) * D_ + h * HD_;
    for (int nf = 0; nf < 4; ++nf)
        for (int r = 0; r < 8; ++r) {
            int row = qrow0 + hi * 8 + r;
            float o = acc[nf][r] / lrun[r];
            cbase[(size_t)row * D_ + nf * 16 + l16] = (__bf16)o;
        }
}

// =============================================================
// Host-side orchestration
// =============================================================
extern "C" void kernel_launch(void* const* d_in, const int* in_sizes, int n_in,
                              void* d_out, int out_size, void* d_ws, size_t ws_size,
                              hipStream_t stream)
{
    const float* x    = (const float*)d_in[0];
    const int*   mask = (const int*)d_in[1];
    const float* Wq   = (const float*)d_in[2];
    const float* bq   = (const float*)d_in[3];
    const float* Wk   = (const float*)d_in[4];
    const float* bk   = (const float*)d_in[5];
    const float* Wv   = (const float*)d_in[6];
    const float* bv   = (const float*)d_in[7];
    const float* Wo   = (const float*)d_in[8];
    const float* bo   = (const float*)d_in[9];
    const float* W1   = (const float*)d_in[10];
    const float* b1   = (const float*)d_in[11];
    const float* W2   = (const float*)d_in[12];
    const float* b2   = (const float*)d_in[13];
    const float* ln1g = (const float*)d_in[14];
    const float* ln1b = (const float*)d_in[15];
    const float* ln2g = (const float*)d_in[16];
    const float* ln2b = (const float*)d_in[17];
    float* out = (float*)d_out;

    char* ws = (char*)d_ws;
    size_t off = 0;
    auto alloc = [&](size_t bytes) -> char* {
        char* p = ws + off;
        off = (off + bytes + 255) & ~(size_t)255;
        return p;
    };
    const size_t DD2  = (size_t)D_ * D_ * 2;
    const size_t DFF2 = (size_t)D_ * FF_ * 2;
    const size_t BSD2 = (size_t)BS_ * D_ * 2;

    __bf16* WqT = (__bf16*)alloc(DD2);
    __bf16* WkT = (__bf16*)alloc(DD2);
    __bf16* WvT = (__bf16*)alloc(DD2);
    __bf16* WoT = (__bf16*)alloc(DD2);
    __bf16* W1T = (__bf16*)alloc(DFF2);
    __bf16* W2T = (__bf16*)alloc(DFF2);
    __bf16* nx  = (__bf16*)alloc(BSD2);
    __bf16* qb  = (__bf16*)alloc(BSD2);
    __bf16* kb  = (__bf16*)alloc(BSD2);
    __bf16* vb  = (__bf16*)alloc(BSD2);
    __bf16* vTb = (__bf16*)alloc(BSD2);
    __bf16* ctx = (__bf16*)alloc(BSD2);
    float*  res = (float*)alloc((size_t)BS_ * D_ * 4);
    __bf16* nr  = (__bf16*)alloc(BSD2);
    __bf16* h1  = (__bf16*)alloc((size_t)BS_ * FF_ * 2);
    (void)ws_size; (void)in_sizes; (void)n_in; (void)out_size;

    // 1. weight conversion to bf16 Bt layouts
    int gDD = (int)(((size_t)D_ * D_) / 256);
    cvt_perhead_kernel<<<gDD, 256, 0, stream>>>(Wq, WqT);
    cvt_perhead_kernel<<<gDD, 256, 0, stream>>>(Wk, WkT);
    cvt_perhead_kernel<<<gDD, 256, 0, stream>>>(Wv, WvT);
    cvt_T_kernel<<<gDD, 256, 0, stream>>>(Wo, WoT, D_, D_);
    int gDFF = (int)(((size_t)D_ * FF_) / 256);
    cvt_T_kernel<<<gDFF, 256, 0, stream>>>(W1, W1T, D_, FF_);   // [FF][D]
    cvt_T_kernel<<<gDFF, 256, 0, stream>>>(W2, W2T, FF_, D_);   // [D][FF]

    // 2. LN1
    ln_kernel<<<BS_, 256, 0, stream>>>(x, ln1g, ln1b, nx);

    // 3. Q/K/V projections (M=4096, N=1024, K=1024)
    dim3 gProj(D_ / 128, BS_ / 128);
    gemm_kernel<false, false, true><<<gProj, 256, 0, stream>>>(
        nx, WqT, bq, nullptr, qb, BS_, D_, D_);
    gemm_kernel<false, false, true><<<gProj, 256, 0, stream>>>(
        nx, WkT, bk, nullptr, kb, BS_, D_, D_);
    gemm_kernel<false, false, true><<<gProj, 256, 0, stream>>>(
        nx, WvT, bv, nullptr, vb, BS_, D_, D_);

    // 4. transpose V -> [B,H,HD,S]
    transpose_v_kernel<<<(int)(((size_t)BS_ * D_) / 256), 256, 0, stream>>>(vb, vTb);

    // 5. flash attention
    dim3 gAttn(S_ / 128, H_, B_);
    attn_kernel<<<gAttn, 256, 0, stream>>>(qb, kb, vTb, mask, ctx);

    // 6. output projection + residual (x)
    gemm_kernel<false, true, false><<<gProj, 256, 0, stream>>>(
        ctx, WoT, bo, x, res, BS_, D_, D_);

    // 7. LN2
    ln_kernel<<<BS_, 256, 0, stream>>>(res, ln2g, ln2b, nr);

    // 8. FFN up + ReLU (N=FF, K=D)
    dim3 gFF1(FF_ / 128, BS_ / 128);
    gemm_kernel<true, false, true><<<gFF1, 256, 0, stream>>>(
        nr, W1T, b1, nullptr, h1, BS_, FF_, D_);

    // 9. FFN down + residual -> fp32 output (N=D, K=FF)
    gemm_kernel<false, true, false><<<gProj, 256, 0, stream>>>(
        h1, W2T, b2, res, out, BS_, D_, FF_);
}
